// GNN_61735859913512
// MI455X (gfx1250) — matrix operand order repaired and checked by reading, not compiled, not run on hardware
//
#include <hip/hip_runtime.h>

// ---------------- problem constants (match reference) ----------------
#define N_NODES 50000
#define N_EDGES 800000
#define GDIM    256
#define H1DIM   128
#define RREL    8
#define HEADS   4
#define H2DIM   64
#define DDIM    256                   // HEADS*H2
#define C1      (RREL*H1DIM + H1DIM)  // 1152: [rel0..rel7 | root]
#define C2      (4*DDIM)              // 1024: [q | k | v | skip]

typedef __attribute__((ext_vector_type(16))) __bf16 v16bf;
typedef __attribute__((ext_vector_type(8)))  float  v8f;

union frag16 { v16bf v; float4 f4[2]; };   // 32 bytes: two b128 halves

// ---------------- helpers ----------------
__device__ __forceinline__ unsigned fenc(float f) {   // order-preserving float->uint
    unsigned u = __float_as_uint(f);
    return (u & 0x80000000u) ? ~u : (u | 0x80000000u);
}
__device__ __forceinline__ float fdec(unsigned u) {
    return (u & 0x80000000u) ? __uint_as_float(u & 0x7fffffffu)
                             : __uint_as_float(~u);
}

__global__ void k_zero_f32(float* p, long n) {
    long i = (long)blockIdx.x * blockDim.x + threadIdx.x;
    if (i < n) p[i] = 0.0f;
}
__global__ void k_fill_u32(unsigned* p, unsigned v, long n) {
    long i = (long)blockIdx.x * blockDim.x + threadIdx.x;
    if (i < n) p[i] = v;
}

// ---------------- pack / cast kernels ----------------
__global__ void k_cast_x(const float* __restrict__ x, __bf16* __restrict__ xb, long n) {
    long i = (long)blockIdx.x * blockDim.x + threadIdx.x;
    if (i < n) xb[i] = (__bf16)x[i];
}

// Weights are packed into WMMA-B fragment order:
//   Bp[ (slab s = k/32) * Nc + col ][ k % 32 ]   (each col's 32 k-values contiguous)
// so a lane's B fragment (K = h*8 .. +7 and 16+h*8 .. +7 of one column) is two 16-byte runs.
__global__ void k_pack_w1(const float* __restrict__ W_rel, const float* __restrict__ W_root,
                          __bf16* __restrict__ Bp) {
    long i = (long)blockIdx.x * blockDim.x + threadIdx.x;
    if (i >= (long)GDIM * C1) return;
    int g = (int)(i / C1), c = (int)(i % C1);          // g = k-row, c = column
    float v;
    if (c < RREL * H1DIM) {
        int r = c / H1DIM, h = c % H1DIM;
        v = W_rel[(long)r * GDIM * H1DIM + (long)g * H1DIM + h];
    } else {
        v = W_root[(long)g * H1DIM + (c - RREL * H1DIM)];
    }
    Bp[((long)(g >> 5) * C1 + c) * 32 + (g & 31)] = (__bf16)v;
}
__global__ void k_pack_w2(const float* __restrict__ Wq, const float* __restrict__ bq,
                          const float* __restrict__ Wk, const float* __restrict__ bk,
                          const float* __restrict__ Wv, const float* __restrict__ bv,
                          const float* __restrict__ Ws, const float* __restrict__ bs,
                          __bf16* __restrict__ Bp, float* __restrict__ biascat) {
    long i = (long)blockIdx.x * blockDim.x + threadIdx.x;
    if (i >= (long)H1DIM * C2) return;
    int h = (int)(i / C2), c = (int)(i % C2);
    int which = c / DDIM, d = c % DDIM;
    const float* W = (which == 0) ? Wq : (which == 1) ? Wk : (which == 2) ? Wv : Ws;
    const float* B = (which == 0) ? bq : (which == 1) ? bk : (which == 2) ? bv : bs;
    Bp[((long)(h >> 5) * C2 + c) * 32 + (h & 31)] = (__bf16)W[(long)h * DDIM + d];
    if (h == 0) biascat[c] = B[d];
}

// ---------------- WMMA GEMM: C[M,Nc] = A[M,K](bf16,row-major) x Bp(packed) + bias ----
// One wave per block; wave computes a 16x64 strip: A fragment loaded once per K-slab
// (2 x b128, contiguous by ISA A-layout), reused across 4 B fragments (2 x b128 each,
// contiguous thanks to pack order). Inner loop = 10 vector loads + 4 WMMAs, zero LDS.
__global__ __launch_bounds__(32)
void k_gemm_bf16(const __bf16* __restrict__ A, const __bf16* __restrict__ Bp,
                 const float* __restrict__ bias, float* __restrict__ C,
                 int M, int K, int Nc) {
    const int lane = threadIdx.x & 31;
    const int half = lane >> 4;
    const int l16  = lane & 15;
    const int m0 = blockIdx.y * 16;
    const int n0 = blockIdx.x * 64;

    v8f acc[4] = {v8f{}, v8f{}, v8f{}, v8f{}};
    const __bf16* arow = A + (long)(m0 + l16) * K + half * 8;

    for (int kc = 0; kc < K; kc += 32) {
        frag16 af;
        af.f4[0] = *(const float4*)(arow + kc);        // K = kc + h*8 .. +7
        af.f4[1] = *(const float4*)(arow + kc + 16);   // K = kc+16+h*8 .. +7
        const __bf16* bslab = Bp + ((long)(kc >> 5) * Nc) * 32 + half * 8;
        #pragma unroll
        for (int t = 0; t < 4; ++t) {
            frag16 bf_;
            const __bf16* bp = bslab + (long)(n0 + 16 * t + l16) * 32;
            bf_.f4[0] = *(const float4*)(bp);          // K = h*8 .. +7 of this column
            bf_.f4[1] = *(const float4*)(bp + 16);     // K = 16+h*8 .. +7
            acc[t] = __builtin_amdgcn_wmma_f32_16x16x32_bf16(
                false, af.v, false, bf_.v, (short)0, acc[t], false, false);
        }
    }
    #pragma unroll
    for (int t = 0; t < 4; ++t) {
        int col = n0 + 16 * t + l16;
        float b = bias ? bias[col] : 0.0f;
        #pragma unroll
        for (int j = 0; j < 8; ++j) {   // C layout: VGPR j, lane l -> M=j+8*half, N=l16
            int row = m0 + j + 8 * half;
            C[(long)row * Nc + col] = acc[t][j] + b;
        }
    }
}

// ---------------- RGCN aggregation ----------------
__global__ void k_count(const int* __restrict__ ei, const int* __restrict__ etype,
                        float* __restrict__ cnt) {
    long e = (long)blockIdx.x * blockDim.x + threadIdx.x;
    if (e >= N_EDGES) return;
    int dst = ei[N_EDGES + e];
    atomicAdd(&cnt[(long)etype[e] * N_NODES + dst], 1.0f);
}
// per (edge, 4-float group): xacc[dst] += h_rel[et][src] / cnt[et,dst]
__global__ void k_scatter_mean(const int* __restrict__ ei, const int* __restrict__ etype,
                               const float* __restrict__ Y1, const float* __restrict__ cnt,
                               float* __restrict__ xacc) {
    long i = (long)blockIdx.x * blockDim.x + threadIdx.x;
    if (i >= (long)N_EDGES * (H1DIM / 4)) return;
    long e = i >> 5;           // H1DIM/4 = 32 groups
    int  g = (int)(i & 31);
    int src = ei[e], dst = ei[N_EDGES + e], et = etype[e];
    const float4 m = *(const float4*)(Y1 + (long)src * C1 + et * H1DIM + g * 4);
    float inv = 1.0f / cnt[(long)et * N_NODES + dst];
    float* o = xacc + (long)dst * H1DIM + g * 4;
    atomicAdd(o + 0, m.x * inv); atomicAdd(o + 1, m.y * inv);
    atomicAdd(o + 2, m.z * inv); atomicAdd(o + 3, m.w * inv);
}
__global__ void k_finish_x(const float* __restrict__ xacc, const float* __restrict__ Y1,
                           const float* __restrict__ b_rgcn, __bf16* __restrict__ xbf) {
    long i = (long)blockIdx.x * blockDim.x + threadIdx.x;
    if (i >= (long)N_NODES * H1DIM) return;
    long n = i / H1DIM; int h = (int)(i % H1DIM);
    float v = xacc[i] + Y1[n * C1 + RREL * H1DIM + h] + b_rgcn[h];
    xbf[i] = (__bf16)v;
}

// ---------------- attention ----------------
__global__ void k_logits_max(const int* __restrict__ ei, const float* __restrict__ qkvs,
                             float* __restrict__ logits, unsigned* __restrict__ mx) {
    long i = (long)blockIdx.x * blockDim.x + threadIdx.x;
    if (i >= (long)N_EDGES * HEADS) return;
    long e = i >> 2; int hd = (int)(i & 3);
    int src = ei[e], dst = ei[N_EDGES + e];
    const float4* qp = (const float4*)(qkvs + (long)dst * C2 + hd * H2DIM);
    const float4* kp = (const float4*)(qkvs + (long)src * C2 + DDIM + hd * H2DIM);
    float acc = 0.0f;
    #pragma unroll
    for (int j = 0; j < H2DIM / 4; ++j) {
        float4 a = qp[j], b = kp[j];
        acc += a.x * b.x + a.y * b.y + a.z * b.z + a.w * b.w;
    }
    acc *= 0.125f;                       // 1/sqrt(64)
    logits[i] = acc;
    atomicMax(&mx[(long)dst * HEADS + hd], fenc(acc));
}
__global__ void k_exp_denom(const int* __restrict__ ei, const unsigned* __restrict__ mx,
                            float* __restrict__ logits, float* __restrict__ denom) {
    long i = (long)blockIdx.x * blockDim.x + threadIdx.x;
    if (i >= (long)N_EDGES * HEADS) return;
    long e = i >> 2; int hd = (int)(i & 3);
    int dst = ei[N_EDGES + e];
    float p = __expf(logits[i] - fdec(mx[(long)dst * HEADS + hd]));
    logits[i] = p;
    atomicAdd(&denom[(long)dst * HEADS + hd], p);
}
__global__ void k_aggregate(const int* __restrict__ ei, const float* __restrict__ logits,
                            const float* __restrict__ denom, const float* __restrict__ qkvs,
                            float* __restrict__ agg) {
    long i = (long)blockIdx.x * blockDim.x + threadIdx.x;
    if (i >= (long)N_EDGES * DDIM) return;
    long e = i >> 8; int d = (int)(i & 255); int hd = d >> 6;
    int src = ei[e], dst = ei[N_EDGES + e];
    float a = logits[e * HEADS + hd] / denom[(long)dst * HEADS + hd];
    atomicAdd(&agg[(long)dst * DDIM + d], a * qkvs[(long)src * C2 + 2 * DDIM + d]);
}
__global__ void k_pre_bn(const float* __restrict__ agg, const float* __restrict__ qkvs,
                         float* __restrict__ pre) {
    long i = (long)blockIdx.x * blockDim.x + threadIdx.x;
    if (i >= (long)N_NODES * DDIM) return;
    long n = i >> 8; int d = (int)(i & 255);
    pre[i] = agg[i] + qkvs[n * C2 + 3 * DDIM + d];    // skip already carries bskip
}

// ---------------- batchnorm + leaky ----------------
#define BN_ROWS 64
__global__ __launch_bounds__(DDIM)
void k_bn_partial(const float* __restrict__ pre, float* __restrict__ bnsum,
                  float* __restrict__ bnsq) {
    int t = threadIdx.x;                 // column
    long base = (long)blockIdx.x * BN_ROWS;
    float s = 0.0f, ss = 0.0f;
    for (int r = 0; r < BN_ROWS; ++r) {
        long row = base + r;
        if (row < N_NODES) { float v = pre[row * DDIM + t]; s += v; ss += v * v; }
    }
    atomicAdd(&bnsum[t], s);
    atomicAdd(&bnsq[t], ss);
}
__global__ void k_bn_final(const float* __restrict__ bnsum, const float* __restrict__ bnsq,
                           float* __restrict__ mu, float* __restrict__ rstd) {
    int d = threadIdx.x;
    if (d >= DDIM) return;
    float m = bnsum[d] / (float)N_NODES;
    float var = bnsq[d] / (float)N_NODES - m * m;
    mu[d] = m;
    rstd[d] = rsqrtf(var + 1e-5f);
}
__global__ void k_bn_apply(const float* __restrict__ pre, const float* __restrict__ mu,
                           const float* __restrict__ rstd, const float* __restrict__ gamma,
                           const float* __restrict__ beta, float* __restrict__ out) {
    long i = (long)blockIdx.x * blockDim.x + threadIdx.x;
    if (i >= (long)N_NODES * DDIM) return;
    int d = (int)(i & 255);
    float y = (pre[i] - mu[d]) * rstd[d] * gamma[d] + beta[d];
    out[i] = (y >= 0.0f) ? y : 0.01f * y;
}

// ---------------- host ----------------
static inline long cdiv(long a, long b) { return (a + b - 1) / b; }

extern "C" void kernel_launch(void* const* d_in, const int* in_sizes, int n_in,
                              void* d_out, int out_size, void* d_ws, size_t ws_size,
                              hipStream_t stream) {
    const float* nf     = (const float*)d_in[0];
    const int*   ei     = (const int*)d_in[2];     // [2,E]: src row, dst row
    const int*   etype  = (const int*)d_in[3];
    const float* W_rel  = (const float*)d_in[4];
    const float* W_root = (const float*)d_in[5];
    const float* b_rgcn = (const float*)d_in[6];
    const float* Wq = (const float*)d_in[7],  *bq = (const float*)d_in[8];
    const float* Wk = (const float*)d_in[9],  *bk = (const float*)d_in[10];
    const float* Wv = (const float*)d_in[11], *bv = (const float*)d_in[12];
    const float* Ws = (const float*)d_in[13], *bs = (const float*)d_in[14];
    const float* gamma = (const float*)d_in[15], *beta = (const float*)d_in[16];
    float* out = (float*)d_out;

    // ---- workspace layout (aliased; peak ~365 MB) ----
    char* base = (char*)d_ws;
    size_t off = 0;
    auto alloc = [&](size_t bytes) -> char* {
        off = (off + 255) & ~(size_t)255;
        char* p = base + off; off += bytes; return p;
    };
    __bf16* Xb     = (__bf16*)alloc((size_t)N_NODES * GDIM * 2);       // dead after GEMM1
    float*  xacc   = (float*) alloc((size_t)N_NODES * H1DIM * 4);      // dead after finish_x
    float*  preBN  = (float*)Xb;                                        // alias: 51.2MB over Xb+xacc
    float*  Y1     = (float*) alloc((size_t)N_NODES * C1 * 4);         // dead after finish_x
    float*  qkvs   = Y1;                                                // alias (204.8 <= 230.4 MB)
    __bf16* Wcat1  = (__bf16*)alloc((size_t)GDIM * C1 * 2);
    __bf16* Wcat2  = (__bf16*)alloc((size_t)H1DIM * C2 * 2);
    float*  bias2  = (float*) alloc((size_t)C2 * 4);
    float*  cnt    = (float*) alloc((size_t)RREL * N_NODES * 4);
    __bf16* xbf    = (__bf16*)alloc((size_t)N_NODES * H1DIM * 2);
    float*  logits = (float*) alloc((size_t)N_EDGES * HEADS * 4);
    unsigned* mx   = (unsigned*)alloc((size_t)N_NODES * HEADS * 4);
    float*  denom  = (float*) alloc((size_t)N_NODES * HEADS * 4);
    float*  agg    = (float*) alloc((size_t)N_NODES * DDIM * 4);
    float*  bnsum  = (float*) alloc((size_t)DDIM * 4);
    float*  bnsq   = (float*) alloc((size_t)DDIM * 4);
    float*  muv    = (float*) alloc((size_t)DDIM * 4);
    float*  rstd   = (float*) alloc((size_t)DDIM * 4);
    (void)ws_size; (void)in_sizes; (void)n_in; (void)out_size;

    const int T = 256;

    // 1) casts / weight packing (weights -> WMMA-B fragment order)
    k_cast_x<<<cdiv((long)N_NODES * GDIM, T), T, 0, stream>>>(nf, Xb, (long)N_NODES * GDIM);
    k_pack_w1<<<cdiv((long)GDIM * C1, T), T, 0, stream>>>(W_rel, W_root, Wcat1);
    k_pack_w2<<<cdiv((long)H1DIM * C2, T), T, 0, stream>>>(Wq, bq, Wk, bk, Wv, bv, Ws, bs,
                                                           Wcat2, bias2);
    // 2) RGCN GEMM: Y1[N,1152] = Xb x Wcat1  (29.5 GFLOP on v_wmma_f32_16x16x32_bf16)
    k_gemm_bf16<<<dim3(C1 / 64, N_NODES / 16), 32, 0, stream>>>(Xb, Wcat1, nullptr, Y1,
                                                                N_NODES, GDIM, C1);
    // 3) per-(relation,dst) counts and mean-scatter
    k_zero_f32<<<cdiv((long)RREL * N_NODES, T), T, 0, stream>>>(cnt, (long)RREL * N_NODES);
    k_zero_f32<<<cdiv((long)N_NODES * H1DIM, T), T, 0, stream>>>(xacc, (long)N_NODES * H1DIM);
    k_count<<<cdiv((long)N_EDGES, T), T, 0, stream>>>(ei, etype, cnt);
    k_scatter_mean<<<cdiv((long)N_EDGES * (H1DIM / 4), T), T, 0, stream>>>(ei, etype, Y1, cnt, xacc);
    k_finish_x<<<cdiv((long)N_NODES * H1DIM, T), T, 0, stream>>>(xacc, Y1, b_rgcn, xbf);
    // 4) q/k/v/skip GEMM: qkvs[N,1024] = xbf x Wcat2 + bias (13.1 GFLOP)
    k_gemm_bf16<<<dim3(C2 / 64, N_NODES / 16), 32, 0, stream>>>(xbf, Wcat2, bias2, qkvs,
                                                                N_NODES, H1DIM, C2);
    // 5) attention: logits -> segment max -> exp/denom -> aggregate
    k_fill_u32<<<cdiv((long)N_NODES * HEADS, T), T, 0, stream>>>(mx, 0x007FFFFFu /*enc(-inf)*/,
                                                                 (long)N_NODES * HEADS);
    k_zero_f32<<<cdiv((long)N_NODES * HEADS, T), T, 0, stream>>>(denom, (long)N_NODES * HEADS);
    k_logits_max<<<cdiv((long)N_EDGES * HEADS, T), T, 0, stream>>>(ei, qkvs, logits, mx);
    k_exp_denom<<<cdiv((long)N_EDGES * HEADS, T), T, 0, stream>>>(ei, mx, logits, denom);
    k_zero_f32<<<cdiv((long)N_NODES * DDIM, T), T, 0, stream>>>(agg, (long)N_NODES * DDIM);
    k_aggregate<<<cdiv((long)N_EDGES * DDIM, T), T, 0, stream>>>(ei, logits, denom, qkvs, agg);
    // 6) skip add, batchnorm, leaky relu
    k_pre_bn<<<cdiv((long)N_NODES * DDIM, T), T, 0, stream>>>(agg, qkvs, preBN);
    k_zero_f32<<<1, 2 * DDIM, 0, stream>>>(bnsum, 2L * DDIM);   // bnsum & bnsq adjacent
    k_bn_partial<<<cdiv((long)N_NODES, BN_ROWS), DDIM, 0, stream>>>(preBN, bnsum, bnsq);
    k_bn_final<<<1, DDIM, 0, stream>>>(bnsum, bnsq, muv, rstd);
    k_bn_apply<<<cdiv((long)N_NODES * DDIM, T), T, 0, stream>>>(preBN, muv, rstd, gamma, beta, out);
}
